// AnomalyMoE_18726057411067
// MI455X (gfx1250) — matrix operand order repaired and checked by reading, not compiled, not run on hardware
//
#include <hip/hip_runtime.h>
#include <math.h>

typedef __attribute__((ext_vector_type(16))) _Float16 v16h;
typedef __attribute__((ext_vector_type(8)))  _Float16 v8h;
typedef __attribute__((ext_vector_type(8)))  float    v8f;

#define NB   16      // batch
#define ND   768     // embed dim
#define NP   784     // patches (28x28)
#define NM   8       // components
#define NE   18      // total experts
#define NTOK 793     // 784 + 1 + 8

// Packed-weight layout (halves), per expert
#define PK_L1   0
#define PK_L2   196608
#define PK_L3   212992
#define PK_L4   217088
#define PK_L5   221184
#define PK_L6   237568
#define PK_EXP  434176   // per-expert stride (halves)
#define PK_GRP  ((size_t)6 * PK_EXP)   // per-group stride (halves)

// ---------------------------------------------------------------------------
// Gating
// ---------------------------------------------------------------------------
__global__ __launch_bounds__(256)
void gate_kernel(const float* __restrict__ cls,
                 const float* __restrict__ gw1, const float* __restrict__ gb1,
                 const float* __restrict__ gw2, const float* __restrict__ gb2,
                 const int*   __restrict__ topk,
                 float* __restrict__ routing) {
  __shared__ float xs[ND];
  __shared__ float hs[256];
  __shared__ float lg[NE];
  int b = blockIdx.x, t = threadIdx.x;
  for (int i = t; i < ND; i += 256) xs[i] = cls[(size_t)b * ND + i];
  __syncthreads();
  float acc = gb1[t];
  for (int k = 0; k < ND; ++k) acc += xs[k] * gw1[(size_t)k * 256 + t];
  hs[t] = acc > 0.f ? acc : 0.f;
  __syncthreads();
  if (t < NE) {
    float a2 = gb2[t];
    for (int k = 0; k < 256; ++k) a2 += hs[k] * gw2[(size_t)k * NE + t];
    lg[t] = a2;
  }
  __syncthreads();
  if (t == 0) {
    int K = topk[0]; if (K > NE) K = NE; if (K < 1) K = 1;
    int used[NE]; float vals[NE]; int idxs[NE];
    for (int i = 0; i < NE; ++i) used[i] = 0;
    for (int j = 0; j < K; ++j) {
      float best = -INFINITY; int bi = 0;
      for (int i = 0; i < NE; ++i)
        if (!used[i] && lg[i] > best) { best = lg[i]; bi = i; }
      used[bi] = 1; vals[j] = best; idxs[j] = bi;
    }
    float mx = vals[0], s = 0.f;
    for (int j = 0; j < K; ++j) { vals[j] = __expf(vals[j] - mx); s += vals[j]; }
    for (int i = 0; i < NE; ++i) routing[b * NE + i] = 0.f;
    for (int j = 0; j < K; ++j) routing[b * NE + idxs[j]] = vals[j] / s;
  }
}

// ---------------------------------------------------------------------------
// Mask downsample + 5x5 dilate + area
// ---------------------------------------------------------------------------
__global__ __launch_bounds__(256)
void mask_kernel(const float* __restrict__ masks,
                 float* __restrict__ m_dil, float* __restrict__ area) {
  __shared__ float ms[NP];
  __shared__ float red[256];
  int bm = blockIdx.x, t = threadIdx.x;
  const float* Mp = masks + (size_t)bm * 224 * 224;
  for (int p = t; p < NP; p += 256) {
    int h = p / 28, w = p % 28;
    ms[p] = Mp[(size_t)(h * 8) * 224 + w * 8];
  }
  __syncthreads();
  float asum = 0.f;
  for (int p = t; p < NP; p += 256) {
    int h = p / 28, w = p % 28;
    float mx = 0.f;
    for (int dh = -2; dh <= 2; ++dh) {
      int hh = h + dh; if (hh < 0 || hh >= 28) continue;
      for (int dw = -2; dw <= 2; ++dw) {
        int ww = w + dw; if (ww < 0 || ww >= 28) continue;
        float v = ms[hh * 28 + ww]; mx = v > mx ? v : mx;
      }
    }
    m_dil[(size_t)bm * NP + p] = mx;
    asum += mx;
  }
  red[t] = asum; __syncthreads();
  for (int s = 128; s > 0; s >>= 1) { if (t < s) red[t] += red[t + s]; __syncthreads(); }
  if (t == 0) area[bm] = red[0] + 1e-8f;
}

// ---------------------------------------------------------------------------
// comp_feat[b,m,d] = sum_p feat[b,d,p] * m_dil[b,m,p] / area[b,m]
// ---------------------------------------------------------------------------
__global__ __launch_bounds__(128)
void comp_kernel(const float* __restrict__ feat, const float* __restrict__ m_dil,
                 const float* __restrict__ area, float* __restrict__ comp_feat) {
  __shared__ float md[NM * NP];
  int b = blockIdx.x, dg = blockIdx.y, t = threadIdx.x;
  for (int i = t; i < NM * NP; i += 128) md[i] = m_dil[(size_t)b * NM * NP + i];
  __syncthreads();
  int d = dg * 128 + t;
  float acc[NM];
#pragma unroll
  for (int m = 0; m < NM; ++m) acc[m] = 0.f;
  const float* F = feat + ((size_t)b * ND + d) * NP;
  for (int p = 0; p < NP; ++p) {
    float f = F[p];
#pragma unroll
    for (int m = 0; m < NM; ++m) acc[m] += f * md[m * NP + p];
  }
  for (int m = 0; m < NM; ++m)
    comp_feat[((size_t)b * NM + m) * ND + d] = acc[m] / area[b * NM + m];
}

// ---------------------------------------------------------------------------
// Weight packer: f32 row-major [IN][OUT] -> f16 WMMA-B lane vectors
// packed[((kt*NT + nt)*32 + lane)*16 + i] = W[(kt*32 + i + (lane&16?16:0))*OUT + nt*16 + (lane&15)]
// ---------------------------------------------------------------------------
__global__ __launch_bounds__(256)
void pack_kernel(const float* __restrict__ W, size_t wStride, int IN, int OUT,
                 _Float16* __restrict__ dst, size_t dStride, size_t layerOff) {
  int KT = IN >> 5, NT = OUT >> 4;
  int total = 6 * KT * NT * 32;
  int tid = blockIdx.x * 256 + threadIdx.x;
  if (tid >= total) return;
  int lane = tid & 31;
  int rest = tid >> 5;
  int nt = rest % NT; rest /= NT;
  int kt = rest % KT; int e = rest / KT;
  const float* Ws = W + (size_t)e * wStride;
  _Float16* d = dst + (size_t)e * dStride + layerOff + ((size_t)(kt * NT + nt) * 32 + lane) * 16;
  int col = nt * 16 + (lane & 15);
  int k0  = kt * 32 + ((lane & 16) ? 16 : 0);
#pragma unroll
  for (int i = 0; i < 16; ++i) d[i] = (_Float16)Ws[(size_t)(k0 + i) * OUT + col];
}

// ---------------------------------------------------------------------------
// WMMA operand loaders
// ---------------------------------------------------------------------------
__device__ __forceinline__ v16h load_A(const _Float16* row, int kbase, int lane) {
  int off = kbase + ((lane & 16) ? 16 : 0);
  const v8h* p0 = (const v8h*)(row + off);
  const v8h* p1 = (const v8h*)(row + off + 8);
  v8h lo = *p0, hi = *p1;
  v16h a;
#pragma unroll
  for (int i = 0; i < 8; ++i) { a[i] = lo[i]; a[i + 8] = hi[i]; }
  return a;
}

__device__ __forceinline__ v16h load_Bp(const _Float16* Wp, int NT, int kt, int nt, int lane) {
  const v8h* p = (const v8h*)(Wp + ((size_t)(kt * NT + nt) * 32 + lane) * 16);
  v8h lo = p[0], hi = p[1];
  v16h b;
#pragma unroll
  for (int i = 0; i < 8; ++i) { b[i] = lo[i]; b[i + 8] = hi[i]; }
  return b;
}

__device__ __forceinline__ v16h load_Bw(const float* W, int OUT, int kbase, int nt, int lane) {
  int col = nt * 16 + (lane & 15);
  int k0  = kbase + ((lane & 16) ? 16 : 0);
  const float* p = W + (size_t)k0 * OUT + col;
  v16h b;
#pragma unroll
  for (int i = 0; i < 16; ++i) b[i] = (_Float16)p[(size_t)i * OUT];
  return b;
}

// ---------------------------------------------------------------------------
// One AE layer: 16 rows, 4 waves, 4 n-tiles per wave per pass (A reused)
// ---------------------------------------------------------------------------
template <int PACKED>
__device__ void ae_layer(const _Float16* __restrict__ in, int inStride, int IN,
                         _Float16* outb, int outStride, int OUT,
                         const _Float16* __restrict__ Wp, const float* __restrict__ W,
                         const float* __restrict__ bias, int do_relu,
                         int wave, int lane) {
  const _Float16* arow = in + (size_t)(lane & 15) * inStride;
  int NT = OUT >> 4, KT = IN >> 5;
  int perWave = NT >> 2;                       // NT divisible by 4
  for (int j0 = 0; j0 < perWave; j0 += 4) {
    int CH = perWave - j0; if (CH > 4) CH = 4; // block-uniform
    v8f acc[4];
#pragma unroll
    for (int c = 0; c < 4; ++c) acc[c] = (v8f){0.f,0.f,0.f,0.f,0.f,0.f,0.f,0.f};
    for (int kt = 0; kt < KT; ++kt) {
      v16h a = load_A(arow, kt * 32, lane);
#pragma unroll
      for (int c = 0; c < 4; ++c) {
        if (c < CH) {
          int nt = wave + 4 * (j0 + c);
          v16h b = PACKED ? load_Bp(Wp, NT, kt, nt, lane)
                          : load_Bw(W, OUT, kt * 32, nt, lane);
          acc[c] = __builtin_amdgcn_wmma_f32_16x16x32_f16(false, a, false, b,
                                                          (short)0, acc[c], false, false);
        }
      }
    }
#pragma unroll
    for (int c = 0; c < 4; ++c) {
      if (c < CH) {
        int nt = wave + 4 * (j0 + c);
        int col = nt * 16 + (lane & 15);
        float bv = bias[col];
        int mb = (lane & 16) ? 8 : 0;
#pragma unroll
        for (int i = 0; i < 8; ++i) {
          float v = acc[c][i] + bv;
          if (do_relu) v = v > 0.f ? v : 0.f;
          outb[(size_t)(mb + i) * outStride + col] = (_Float16)v;
        }
      }
    }
  }
}

template <int PACKED>
__device__ void ae_layer_final(const _Float16* __restrict__ in, int inStride, int IN,
                               float* outacc, int OUT,
                               const _Float16* __restrict__ Wp, const float* __restrict__ W,
                               const float* __restrict__ bias,
                               const float* wrow, int wave, int lane) {
  const _Float16* arow = in + (size_t)(lane & 15) * inStride;
  int NT = OUT >> 4, KT = IN >> 5;
  int perWave = NT >> 2;
  for (int j0 = 0; j0 < perWave; j0 += 4) {
    int CH = perWave - j0; if (CH > 4) CH = 4;
    v8f acc[4];
#pragma unroll
    for (int c = 0; c < 4; ++c) acc[c] = (v8f){0.f,0.f,0.f,0.f,0.f,0.f,0.f,0.f};
    for (int kt = 0; kt < KT; ++kt) {
      v16h a = load_A(arow, kt * 32, lane);
#pragma unroll
      for (int c = 0; c < 4; ++c) {
        if (c < CH) {
          int nt = wave + 4 * (j0 + c);
          v16h b = PACKED ? load_Bp(Wp, NT, kt, nt, lane)
                          : load_Bw(W, OUT, kt * 32, nt, lane);
          acc[c] = __builtin_amdgcn_wmma_f32_16x16x32_f16(false, a, false, b,
                                                          (short)0, acc[c], false, false);
        }
      }
    }
#pragma unroll
    for (int c = 0; c < 4; ++c) {
      if (c < CH) {
        int nt = wave + 4 * (j0 + c);
        int col = nt * 16 + (lane & 15);
        float bv = bias[col];
        int mb = (lane & 16) ? 8 : 0;
#pragma unroll
        for (int i = 0; i < 8; ++i) {
          int m = mb + i;
          outacc[(size_t)m * OUT + col] += wrow[m] * (acc[c][i] + bv);
        }
      }
    }
  }
}

// ---------------------------------------------------------------------------
// AE tile kernel: task 0=patch, 1=global, 2=comp
// ---------------------------------------------------------------------------
#define SMEM_OUT 0
#define SMEM_X   (16 * 768 * 4)                  // 49152
#define SMEM_P   (SMEM_X + 16 * 784 * 2)         // +25088
#define SMEM_Q   (SMEM_P + 16 * 272 * 2)         // +8704
#define SMEM_W   (SMEM_Q + 16 * 272 * 2)         // +8704
#define SMEM_AE  (SMEM_W + 64)                   // 91712 bytes

template <int PACKED>
__global__ __launch_bounds__(128)
void ae_kernel(int task, int routeBase,
               const float* __restrict__ X, const float* __restrict__ routing,
               const _Float16* __restrict__ Wpk,
               const float* we1, const float* we2, const float* we3,
               const float* wd1, const float* wd2, const float* wd3,
               const float* be1, const float* be2, const float* be3,
               const float* bd1, const float* bd2, const float* bd3,
               float* __restrict__ out) {
  extern __shared__ char smem[];
  float*    outacc = (float*)(smem + SMEM_OUT);
  _Float16* bufX   = (_Float16*)(smem + SMEM_X);
  _Float16* bufP   = (_Float16*)(smem + SMEM_P);
  _Float16* bufQ   = (_Float16*)(smem + SMEM_Q);
  float*    wrow   = (float*)(smem + SMEM_W);
  const int SX = 784, SP = 272;

  int t = threadIdx.x, lane = t & 31, wave = t >> 5;
  int tile = blockIdx.x;

  for (int i = t; i < 16 * ND; i += 128) outacc[i] = 0.f;

  if (task == 0) {
    int b = tile / 49, p0 = (tile % 49) * 16;
    const float* F = X + (size_t)b * ND * NP + p0;
    int pl = t & 15, d0 = t >> 4;
    for (int d = d0; d < ND; d += 8)
      bufX[(size_t)pl * SX + d] = (_Float16)F[(size_t)d * NP + pl];
  } else if (task == 1) {
    for (int i = t; i < 16 * ND; i += 128)
      bufX[(size_t)(i / ND) * SX + (i % ND)] = (_Float16)X[i];
  } else {
    const float* F = X + (size_t)tile * 16 * ND;
    for (int i = t; i < 16 * ND; i += 128)
      bufX[(size_t)(i / ND) * SX + (i % ND)] = (_Float16)F[i];
  }
  __syncthreads();

  for (int e = 0; e < 6; ++e) {
    if (t < 16) {
      int b;
      if (task == 0)      b = tile / 49;
      else if (task == 1) b = t;
      else                b = (tile * 16 + t) >> 3;
      wrow[t] = routing[b * NE + routeBase + e];
    }
    __syncthreads();
    int any = 0;
    for (int r = 0; r < 16; ++r) any |= (wrow[r] != 0.f);
    if (any) {
      const _Float16* Pe = Wpk + (size_t)e * PK_EXP;
      const float* W1 = we1 + (size_t)e * 768 * 256; const float* B1 = be1 + e * 256;
      const float* W2 = we2 + (size_t)e * 256 * 64;  const float* B2 = be2 + e * 64;
      const float* W3 = we3 + (size_t)e * 64 * 64;   const float* B3 = be3 + e * 64;
      const float* W4 = wd1 + (size_t)e * 64 * 64;   const float* B4 = bd1 + e * 64;
      const float* W5 = wd2 + (size_t)e * 64 * 256;  const float* B5 = bd2 + e * 256;
      const float* W6 = wd3 + (size_t)e * 256 * 768; const float* B6 = bd3 + e * 768;
      ae_layer<PACKED>(bufX, SX, 768, bufP, SP, 256, Pe + PK_L1, W1, B1, 1, wave, lane); __syncthreads();
      ae_layer<PACKED>(bufP, SP, 256, bufQ, SP,  64, Pe + PK_L2, W2, B2, 1, wave, lane); __syncthreads();
      ae_layer<PACKED>(bufQ, SP,  64, bufP, SP,  64, Pe + PK_L3, W3, B3, 0, wave, lane); __syncthreads();
      ae_layer<PACKED>(bufP, SP,  64, bufQ, SP,  64, Pe + PK_L4, W4, B4, 1, wave, lane); __syncthreads();
      ae_layer<PACKED>(bufQ, SP,  64, bufP, SP, 256, Pe + PK_L5, W5, B5, 1, wave, lane); __syncthreads();
      ae_layer_final<PACKED>(bufP, SP, 256, outacc, 768, Pe + PK_L6, W6, B6, wrow, wave, lane);
    }
    __syncthreads();
  }

  for (int i = t; i < 16 * ND; i += 128) {
    int r = i / ND, col = i % ND;
    size_t addr;
    if (task == 0) {
      int b = tile / 49, p = (tile % 49) * 16 + r;
      addr = ((size_t)b * NTOK + p) * ND + col;
    } else if (task == 1) {
      addr = ((size_t)r * NTOK + NP) * ND + col;
    } else {
      int g = tile * 16 + r;
      addr = ((size_t)(g >> 3) * NTOK + NP + 1 + (g & 7)) * ND + col;
    }
    out[addr] = outacc[i];
  }
}

// ---------------------------------------------------------------------------
extern "C" void kernel_launch(void* const* d_in, const int* in_sizes, int n_in,
                              void* d_out, int out_size, void* d_ws, size_t ws_size,
                              hipStream_t stream) {
  const float* cls   = (const float*)d_in[0];
  const float* feat  = (const float*)d_in[1];
  const float* gfeat = (const float*)d_in[2];
  const float* masks = (const float*)d_in[3];
  const float* gw1   = (const float*)d_in[4];
  const float* gb1   = (const float*)d_in[5];
  const float* gw2   = (const float*)d_in[6];
  const float* gb2   = (const float*)d_in[7];
  const float* pe[12]; const float* ge[12]; const float* ce[12];
  for (int i = 0; i < 12; ++i) {
    pe[i] = (const float*)d_in[8 + i];
    ge[i] = (const float*)d_in[20 + i];
    ce[i] = (const float*)d_in[32 + i];
  }
  const int* topk = (const int*)d_in[44];

  float* ws        = (float*)d_ws;
  float* routing   = ws;                            // 512 floats (16*18 used)
  float* m_dil     = ws + 512;                      // 16*8*784
  float* area      = ws + 512 + NB * NM * NP;       // 128 floats (16*8 used)
  float* comp_feat = area + 128;                    // 16*8*768
  float* out       = (float*)d_out;

  const size_t baseBytes = (512 + (size_t)NB * NM * NP + 128 + (size_t)NB * NM * ND) * 4; // 797184
  const size_t pkBytes   = 3 * PK_GRP * 2;          // 15,630,336
  int packed = (ws_size >= baseBytes + pkBytes) ? 1 : 0;
  _Float16* pkBase = (_Float16*)((char*)d_ws + baseBytes);
  _Float16* pkPE = pkBase;
  _Float16* pkGE = pkBase + PK_GRP;
  _Float16* pkCE = pkBase + 2 * PK_GRP;

  gate_kernel<<<NB, 256, 0, stream>>>(cls, gw1, gb1, gw2, gb2, topk, routing);
  mask_kernel<<<NB * NM, 256, 0, stream>>>(masks, m_dil, area);
  comp_kernel<<<dim3(NB, 6), 128, 0, stream>>>(feat, m_dil, area, comp_feat);

  if (packed) {
    const int IN[6]  = {768, 256, 64, 64, 64, 256};
    const int OUT[6] = {256, 64, 64, 64, 256, 768};
    const size_t LOFF[6] = {PK_L1, PK_L2, PK_L3, PK_L4, PK_L5, PK_L6};
    const int WIDX[6] = {0, 2, 4, 6, 8, 10};
    _Float16* dsts[3] = {pkPE, pkGE, pkCE};
    const float* const* grps[3] = {pe, ge, ce};
    for (int g = 0; g < 3; ++g)
      for (int l = 0; l < 6; ++l) {
        int KT = IN[l] >> 5, NT = OUT[l] >> 4;
        int total = 6 * KT * NT * 32;
        pack_kernel<<<(total + 255) / 256, 256, 0, stream>>>(
            grps[g][WIDX[l]], (size_t)IN[l] * OUT[l], IN[l], OUT[l],
            dsts[g], (size_t)PK_EXP, LOFF[l]);
      }
  }

#define AE_ARGS(grp, pk) \
  (pk), grp[0], grp[2], grp[4], grp[6], grp[8], grp[10], \
        grp[1], grp[3], grp[5], grp[7], grp[9], grp[11], out

  if (packed) {
    ae_kernel<1><<<1,   128, SMEM_AE, stream>>>(1,  6, gfeat,     routing, AE_ARGS(ge, pkGE));
    ae_kernel<1><<<8,   128, SMEM_AE, stream>>>(2, 12, comp_feat, routing, AE_ARGS(ce, pkCE));
    ae_kernel<1><<<784, 128, SMEM_AE, stream>>>(0,  0, feat,      routing, AE_ARGS(pe, pkPE));
  } else {
    ae_kernel<0><<<1,   128, SMEM_AE, stream>>>(1,  6, gfeat,     routing, AE_ARGS(ge, (const _Float16*)0));
    ae_kernel<0><<<8,   128, SMEM_AE, stream>>>(2, 12, comp_feat, routing, AE_ARGS(ce, (const _Float16*)0));
    ae_kernel<0><<<784, 128, SMEM_AE, stream>>>(0,  0, feat,      routing, AE_ARGS(pe, (const _Float16*)0));
  }
#undef AE_ARGS
}
